// BiLSTM_CRF_73143293051466
// MI455X (gfx1250) — compile-verified, matrix operand order checked by reading, not run on hardware
//
#include <hip/hip_runtime.h>
#include <hip/hip_bf16.h>
#include <cstdint>

#define B_  64
#define T_  512
#define E_  128
#define H_  256
#define K_  9
#define G4_ 1024   // 4*H

typedef __attribute__((ext_vector_type(16))) __bf16 v16bf;
typedef __attribute__((ext_vector_type(8)))  float  v8f;

__device__ __forceinline__ v8f wmma_bf16(v16bf a, v16bf b, v8f c) {
  // D(16x16 f32) = A(16x32 bf16) * B(32x16 bf16) + C
  return __builtin_amdgcn_wmma_f32_16x16x32_bf16(false, a, false, b,
                                                 (short)0, c, false, false);
}

// Stop cross-section load CSE/hoisting (keeps per-gate register pressure low
// instead of caching all A fragments across gates and spilling to scratch).
__device__ __forceinline__ void gate_fence() {
  __builtin_amdgcn_sched_barrier(0);
  asm volatile("" ::: "memory");
}

// Generic 16x32 bf16 fragment load (A-layout; also used for B operand).
// Per ISA 7.12.2: lanes 0-15 hold row m=lane, elems 0..7 -> k0..k0+7,
// elems 8..15 -> k0+16..k0+23; lanes 16-31 start at +8.
__device__ __forceinline__ v16bf frag_load(const __bf16* base, long row_stride,
                                           int row0, int k0) {
  const int lane = threadIdx.x & 31;
  const __bf16* rp = base + (long)(row0 + (lane & 15)) * row_stride;
  const int kb = k0 + ((lane >> 4) << 3);
  v16bf v;
#pragma unroll
  for (int e = 0; e < 8; ++e) {
    v[e]     = rp[kb + e];
    v[8 + e] = rp[kb + 16 + e];
  }
  return v;
}

__device__ __forceinline__ float sigmoidf_(float x) {
  return 1.0f / (1.0f + __expf(-x));
}

// ---------------------------------------------------------------- prep ----
__global__ void prep_kernel(const float* __restrict__ Wih_f, const float* __restrict__ Whh_f,
                            const float* __restrict__ bih_f, const float* __restrict__ bhh_f,
                            const float* __restrict__ Wih_b, const float* __restrict__ Whh_b,
                            const float* __restrict__ bih_b, const float* __restrict__ bhh_b,
                            __bf16* __restrict__ Wih, __bf16* __restrict__ Whh,
                            float* __restrict__ bias) {
  const int i = blockIdx.x * blockDim.x + threadIdx.x;
  const int nIh = G4_ * E_;   // 131072
  const int nHh = G4_ * H_;   // 262144
  if (i < nIh) { Wih[i] = (__bf16)Wih_f[i]; Wih[nIh + i] = (__bf16)Wih_b[i]; }
  if (i < nHh) { Whh[i] = (__bf16)Whh_f[i]; Whh[nHh + i] = (__bf16)Whh_b[i]; }
  if (i < G4_) { bias[i] = bih_f[i] + bhh_f[i]; bias[G4_ + i] = bih_b[i] + bhh_b[i]; }
}

// --------------------------------------------------------------- embed ----
__global__ void embed_kernel(const int* __restrict__ sent,
                             const float* __restrict__ emb,
                             __bf16* __restrict__ x) {
  const long i = (long)blockIdx.x * blockDim.x + threadIdx.x;  // over B*T*E
  if (i >= (long)B_ * T_ * E_) return;
  const int e  = (int)(i % E_);
  const long bt = i / E_;
  const int tok = sent[bt];
  x[i] = (__bf16)emb[(long)tok * E_ + e];
}

// ---------------------------------------------------------------- lstm ----
// grid = 2 (dir), block = 1024 (32 waves). Persistent over all T steps.
// Wave w owns j-tile (w & 15) and batch tiles (w>>4) and (w>>4)+2; the two
// batch tiles share identical weight fragments, so B is loaded once per pair.
__global__ __launch_bounds__(1024, 1) void lstm_kernel(
    const __bf16* __restrict__ x,     // [B,T,E]
    const __bf16* __restrict__ Wih,   // [2][4H][E]
    const __bf16* __restrict__ Whh,   // [2][4H][H]
    const float*  __restrict__ bias,  // [2][4H]
    const float*  __restrict__ h0,    // [2,B,H]
    const float*  __restrict__ c0,    // [2,B,H]
    float* __restrict__ hs)           // [2][B][T][H]
{
  __shared__ __bf16 hsh[B_][H_];        // 32 KB recurrent state
  __shared__ __bf16 xsh[2][B_][E_];     // 2 x 16 KB async-staged x_t tiles

  const int dir = blockIdx.x;
  const __bf16* Wi = Wih + (long)dir * G4_ * E_;
  const __bf16* Wh = Whh + (long)dir * G4_ * H_;
  const float*  bs = bias + dir * G4_;
  float* out = hs + (long)dir * B_ * T_ * H_;

  const int tid  = threadIdx.x;
  const int wave = tid >> 5;
  const int lane = tid & 31;
  const int jt   = wave & 15;       // j tile (16 cols)
  const int bt0  = wave >> 4;       // batch tiles 0..1
  const int bt1  = bt0 + 2;         // batch tiles 2..3
  const int j    = jt * 16 + (lane & 15);
  const int rofs = (lane >= 16) ? 8 : 0;

  // async stage of one x_t tile: 1024 threads x 16B = 16 KB
  const int xb = tid >> 4;          // row 0..63
  const int xc = (tid & 15) << 3;   // col chunk (8 bf16)
  auto issue_x = [&](int t, int buf) {
    unsigned long long ga =
        (unsigned long long)(uintptr_t)(x + ((long)xb * T_ + t) * E_ + xc);
    unsigned la = (unsigned)(uintptr_t)(&xsh[buf][xb][xc]);  // low 32b = LDS addr
    asm volatile("global_load_async_to_lds_b128 %0, %1, off"
                 :: "v"(la), "v"(ga) : "memory");
  };

  for (int i = tid; i < B_ * H_; i += 1024)
    hsh[i / H_][i % H_] = (__bf16)h0[dir * B_ * H_ + i];

  // cell state lives in accumulator-layout registers for the whole sequence
  v8f c0f, c1f;
#pragma unroll
  for (int r = 0; r < 8; ++r) {
    c0f[r] = c0[dir * B_ * H_ + (bt0 * 16 + r + rofs) * H_ + j];
    c1f[r] = c0[dir * B_ * H_ + (bt1 * 16 + r + rofs) * H_ + j];
  }

  issue_x(dir ? (T_ - 1) : 0, 0);
  int cur = 0;

  // one gate's pre-activation GEMM for both batch tiles (B shared)
  auto gemm_gate = [&](int gg, v8f& A0, v8f& A1) {
#pragma unroll
    for (int r = 0; r < 8; ++r) { A0[r] = 0.0f; A1[r] = 0.0f; }
    const int n0 = gg * H_ + jt * 16;
    for (int kk = 0; kk < E_; kk += 32) {
      v16bf bfr = frag_load(Wi, E_, n0, kk);
      v16bf a0  = frag_load(&xsh[cur][0][0], E_, bt0 * 16, kk);
      v16bf a1  = frag_load(&xsh[cur][0][0], E_, bt1 * 16, kk);
      A0 = wmma_bf16(a0, bfr, A0);
      A1 = wmma_bf16(a1, bfr, A1);
    }
    for (int kk = 0; kk < H_; kk += 32) {
      v16bf bfr = frag_load(Wh, H_, n0, kk);
      v16bf a0  = frag_load(&hsh[0][0], H_, bt0 * 16, kk);
      v16bf a1  = frag_load(&hsh[0][0], H_, bt1 * 16, kk);
      A0 = wmma_bf16(a0, bfr, A0);
      A1 = wmma_bf16(a1, bfr, A1);
    }
  };

  for (int step = 0; step < T_; ++step) {
    const int t = dir ? (T_ - 1 - step) : step;
    asm volatile("s_wait_asynccnt 0" ::: "memory");
    __syncthreads();                          // xsh[cur] + hsh ready for all
    if (step + 1 < T_) issue_x(dir ? (T_ - 2 - step) : (step + 1), cur ^ 1);

    __builtin_prefetch(Wh, 0, 1);

    v8f acc0, acc1, st0, st1;
    // gate i  -> stash sigmoid(i)
    gemm_gate(0, acc0, acc1);
    {
      const float bb = bs[0 * H_ + j];
#pragma unroll
      for (int r = 0; r < 8; ++r) {
        st0[r] = sigmoidf_(acc0[r] + bb);
        st1[r] = sigmoidf_(acc1[r] + bb);
      }
    }
    gate_fence();
    // gate g  -> stash = sigmoid(i)*tanh(g)
    gemm_gate(2, acc0, acc1);
    {
      const float bb = bs[2 * H_ + j];
#pragma unroll
      for (int r = 0; r < 8; ++r) {
        st0[r] *= tanhf(acc0[r] + bb);
        st1[r] *= tanhf(acc1[r] + bb);
      }
    }
    gate_fence();
    // gate f  -> c = sigmoid(f)*c + stash
    gemm_gate(1, acc0, acc1);
    {
      const float bb = bs[1 * H_ + j];
#pragma unroll
      for (int r = 0; r < 8; ++r) {
        c0f[r] = sigmoidf_(acc0[r] + bb) * c0f[r] + st0[r];
        c1f[r] = sigmoidf_(acc1[r] + bb) * c1f[r] + st1[r];
      }
    }
    gate_fence();
    // gate o  -> h = sigmoid(o)*tanh(c)
    gemm_gate(3, acc0, acc1);
    v8f hn0, hn1;
    {
      const float bb = bs[3 * H_ + j];
#pragma unroll
      for (int r = 0; r < 8; ++r) {
        hn0[r] = sigmoidf_(acc0[r] + bb) * tanhf(c0f[r]);
        hn1[r] = sigmoidf_(acc1[r] + bb) * tanhf(c1f[r]);
      }
    }

    // global h output: race-free, overlap with other waves' GEMMs
#pragma unroll
    for (int r = 0; r < 8; ++r) {
      out[((long)(bt0 * 16 + r + rofs) * T_ + t) * H_ + j] = hn0[r];
      out[((long)(bt1 * 16 + r + rofs) * T_ + t) * H_ + j] = hn1[r];
    }

    __syncthreads();                          // all GEMM reads of hsh done
#pragma unroll
    for (int r = 0; r < 8; ++r) {
      hsh[bt0 * 16 + r + rofs][j] = (__bf16)hn0[r];
      hsh[bt1 * 16 + r + rofs][j] = (__bf16)hn1[r];
    }
    __syncthreads();                          // h updates visible
    cur ^= 1;
  }
}

// --------------------------------------------------------------- feats ----
__global__ void feats_kernel(const float* __restrict__ hs,     // [2][B][T][H]
                             const float* __restrict__ Wout,   // [K,2H]
                             const float* __restrict__ bout,   // [K]
                             float* __restrict__ feats) {      // [B,T,K]
  const long i = (long)blockIdx.x * blockDim.x + threadIdx.x;
  if (i >= (long)B_ * T_ * K_) return;
  const int  k  = (int)(i % K_);
  const long bt = i / K_;
  const float* hf = hs + bt * H_;
  const float* hb = hs + (long)B_ * T_ * H_ + bt * H_;
  const float* w  = Wout + (long)k * 2 * H_;
  float s = bout[k];
  for (int j = 0; j < H_; ++j) s += w[j] * hf[j];
  for (int j = 0; j < H_; ++j) s += w[H_ + j] * hb[j];
  feats[i] = s;
}

// ------------------------------------------------------------- viterbi ----
__global__ void viterbi_kernel(const float* __restrict__ feats,
                               const unsigned char* __restrict__ mask,
                               const float* __restrict__ start,
                               const float* __restrict__ endv,
                               const float* __restrict__ trans,
                               int* __restrict__ hist,    // [B][T-1][K]
                               int* __restrict__ tags) {  // [B][T]
  const int b = blockIdx.x * blockDim.x + threadIdx.x;
  if (b >= B_) return;
  float score[K_];
  for (int k = 0; k < K_; ++k) score[k] = start[k] + feats[(long)b * T_ * K_ + k];
  int* hb = hist + (long)b * (T_ - 1) * K_;
  for (int t = 1; t < T_; ++t) {
    float ns[K_]; int ni[K_];
    for (int kn = 0; kn < K_; ++kn) {
      float best = score[0] + trans[kn];
      int bi = 0;
      for (int kp = 1; kp < K_; ++kp) {
        const float v = score[kp] + trans[kp * K_ + kn];
        if (v > best) { best = v; bi = kp; }
      }
      ns[kn] = best + feats[((long)b * T_ + t) * K_ + kn];
      ni[kn] = bi;
    }
    const bool m = mask[b * T_ + t] != 0;
    for (int kn = 0; kn < K_; ++kn) {
      if (m) score[kn] = ns[kn];
      hb[(t - 1) * K_ + kn] = ni[kn];
    }
  }
  float best = score[0] + endv[0]; int last = 0;
  for (int k = 1; k < K_; ++k) {
    const float v = score[k] + endv[k];
    if (v > best) { best = v; last = k; }
  }
  int cur = last;
  tags[b * T_ + (T_ - 1)] = cur;
  for (int t = T_ - 1; t >= 1; --t) {
    cur = hb[(t - 1) * K_ + cur];
    tags[b * T_ + (t - 1)] = cur;
  }
}

// -------------------------------------------------------------- launch ----
extern "C" void kernel_launch(void* const* d_in, const int* in_sizes, int n_in,
                              void* d_out, int out_size, void* d_ws, size_t ws_size,
                              hipStream_t stream) {
  const int*   sent  = (const int*)  d_in[0];
  const unsigned char* mask = (const unsigned char*)d_in[1];
  const float* emb   = (const float*)d_in[2];
  const float* Wih_f = (const float*)d_in[3];
  const float* Whh_f = (const float*)d_in[4];
  const float* bih_f = (const float*)d_in[5];
  const float* bhh_f = (const float*)d_in[6];
  const float* Wih_b = (const float*)d_in[7];
  const float* Whh_b = (const float*)d_in[8];
  const float* bih_b = (const float*)d_in[9];
  const float* bhh_b = (const float*)d_in[10];
  const float* Wout  = (const float*)d_in[11];
  const float* bout  = (const float*)d_in[12];
  const float* start = (const float*)d_in[13];
  const float* endv  = (const float*)d_in[14];
  const float* trans = (const float*)d_in[15];
  const float* h0    = (const float*)d_in[16];
  const float* c0    = (const float*)d_in[17];
  int* tags = (int*)d_out;

  char* ws = (char*)d_ws;
  size_t off = 0;
  auto walloc = [&](size_t bytes) -> void* {
    void* p = ws + off;
    off = (off + bytes + 255) & ~(size_t)255;
    return p;
  };
  __bf16* x_bf  = (__bf16*)walloc((size_t)B_ * T_ * E_ * sizeof(__bf16));
  __bf16* Wih   = (__bf16*)walloc((size_t)2 * G4_ * E_ * sizeof(__bf16));
  __bf16* Whh   = (__bf16*)walloc((size_t)2 * G4_ * H_ * sizeof(__bf16));
  float*  bias  = (float*) walloc((size_t)2 * G4_ * sizeof(float));
  float*  hsbuf = (float*) walloc((size_t)2 * B_ * T_ * H_ * sizeof(float));
  float*  feats = (float*) walloc((size_t)B_ * T_ * K_ * sizeof(float));
  int*    hist  = (int*)   walloc((size_t)B_ * (T_ - 1) * K_ * sizeof(int));

  // 1) weights -> bf16, fold biases
  prep_kernel<<<(G4_ * H_ + 255) / 256, 256, 0, stream>>>(
      Wih_f, Whh_f, bih_f, bhh_f, Wih_b, Whh_b, bih_b, bhh_b, Wih, Whh, bias);

  // 2) embedding gather + bf16 convert
  {
    const long n = (long)B_ * T_ * E_;
    embed_kernel<<<(unsigned)((n + 255) / 256), 256, 0, stream>>>(sent, emb, x_bf);
  }

  // 3) bidirectional LSTM: one persistent WGP-resident block per direction
  lstm_kernel<<<2, 1024, 0, stream>>>(x_bf, Wih, Whh, bias, h0, c0, hsbuf);

  // 4) output projection
  {
    const long n = (long)B_ * T_ * K_;
    feats_kernel<<<(unsigned)((n + 255) / 256), 256, 0, stream>>>(hsbuf, Wout, bout, feats);
  }

  // 5) CRF Viterbi decode (64 independent chains)
  viterbi_kernel<<<2, 32, 0, stream>>>(feats, mask, start, endv, trans, hist, tags);
}
